// SelfAttention_83502754169246
// MI455X (gfx1250) — matrix operand order repaired and checked by reading, not compiled, not run on hardware
//
#include <hip/hip_runtime.h>
#include <hip/hip_bf16.h>

typedef __attribute__((ext_vector_type(16))) _Float16 v16h;
typedef __attribute__((ext_vector_type(8)))  _Float16 v8h;
typedef __attribute__((ext_vector_type(4)))  _Float16 v4h;
typedef __attribute__((ext_vector_type(8)))  float    v8f;
typedef __attribute__((ext_vector_type(4)))  int      v4i;
typedef __attribute__((ext_vector_type(4)))  unsigned v4u;
typedef __attribute__((ext_vector_type(8)))  unsigned v8u;

#define GLOBAL_AS __attribute__((address_space(1)))
#define LDS_AS    __attribute__((address_space(3)))

// LDS tile row stride: 32 payload halves + 8 pad halves = 80 bytes = 20 banks.
// Matches TDM padding: pad_interval=3 (16 DWORDs = 64B payload) +
// pad_amount=3 (4 DWORDs = 16B pad) -> 80B row pitch.
#define LDS_STRIDE 40
#define TILE_HALFS (128 * LDS_STRIDE)

// ---------------------------------------------------------------------------
// CDNA5 async global->LDS staging (ASYNCcnt-tracked), with safe fallback.
// ---------------------------------------------------------------------------
#if __has_builtin(__builtin_amdgcn_global_load_async_to_lds_b128)
#define ASYNC_LDS 1
#else
#define ASYNC_LDS 0
#endif

__device__ __forceinline__ void wait_async_all() {
#if ASYNC_LDS
#if __has_builtin(__builtin_amdgcn_s_wait_asynccnt)
    __builtin_amdgcn_s_wait_asynccnt(0);
#else
    asm volatile("s_wait_asynccnt 0" ::: "memory");
#endif
#endif
}

__device__ __forceinline__ void wait_tensor_all() {
#if __has_builtin(__builtin_amdgcn_s_wait_tensorcnt)
    __builtin_amdgcn_s_wait_tensorcnt(0);
#else
    asm volatile("s_wait_tensorcnt 0" ::: "memory");
#endif
}

// LDS byte offset of a __shared__ pointer (ptrtoint of AS3 pointer).
__device__ __forceinline__ unsigned lds_offset_of(const _Float16* p) {
    return (unsigned)(unsigned long long)(LDS_AS const _Float16*)p;
}

// ---------------------------------------------------------------------------
// Tensor Data Mover: DMA a 128x32 f16 tile (row-major, leading dim `ld`
// elements) from global into a padded LDS tile (row pitch 80B).
// D# layout per CDNA5 ISA 8.3/8.4; 2-operand form => VADDR2/3 NULL (2-D).
// Issue from ONE wave only (EXEC ignored); drain with s_wait_tensorcnt.
// ---------------------------------------------------------------------------
__device__ __forceinline__ void tdm_load_tile_f16(const _Float16* gsrc,
                                                  unsigned lds_byte_off,
                                                  unsigned ld_elems) {
    const unsigned long long ga = (unsigned long long)gsrc;
    v4u g0;
    g0[0] = 1u;                                   // count=1, user descriptor
    g0[1] = lds_byte_off;                         // lds_addr
    g0[2] = (unsigned)ga;                         // global_addr[31:0]
    g0[3] = (unsigned)((ga >> 32) & 0x01FFFFFFu)  // global_addr[56:32]
          | 0x80000000u;                          // type=2 ("image")
    v8u g1;
    g1[0] = (1u << 16)                            // data_size = 2 bytes
          | (1u << 20)                            // pad_enable
          | (3u << 22)                            // pad_interval: 16 DWORDs (64B)
          | (3u << 25);                           // pad_amount: 4 DWORDs (16B)
    g1[1] = (ld_elems & 0xFFFFu) << 16;           // tensor_dim0[15:0]
    g1[2] = ((ld_elems >> 16) & 0xFFFFu)          // tensor_dim0[31:16]
          | (128u << 16);                         // tensor_dim1[15:0] = 128 rows
    g1[3] = (32u << 16);                          // tensor_dim1 hi=0 | tile_dim0=32
    g1[4] = 128u;                                 // tile_dim1=128, tile_dim2=0
    g1[5] = ld_elems;                             // tensor_dim0_stride[31:0]
    g1[6] = 0u;                                   // stride0 hi | stride1 lo
    g1[7] = 0u;                                   // stride1 hi
    asm volatile("tensor_load_to_lds %0, %1" :: "s"(g0), "s"(g1) : "memory");
}

// Copy a 128x32 f16 tile into LDS: 2 x 16B async chunks per thread.
__device__ __forceinline__ void stage_f16(const _Float16* __restrict__ src, size_t ld,
                                          _Float16* dst, int tid) {
#pragma unroll
    for (int j = 0; j < 2; ++j) {
        const int chunk = tid + j * 256;          // 512 chunks of 8 halves
        const int r  = chunk >> 2;
        const int c8 = (chunk & 3) * 8;
        const _Float16* g = src + (size_t)r * ld + c8;
        _Float16*       l = dst + r * LDS_STRIDE + c8;
#if ASYNC_LDS
        __builtin_amdgcn_global_load_async_to_lds_b128(
            (GLOBAL_AS v4i*)const_cast<_Float16*>(g),
            (LDS_AS v4i*)l, 0, 0);
#else
        *(v8h*)l = *(const v8h*)g;
#endif
    }
}

// Copy a 128x32 f32 tile, converting to f16 (b128 loads, b64 LDS stores).
__device__ __forceinline__ void stage_f32_cvt(const float* __restrict__ src, size_t ld,
                                              _Float16* dst, int tid) {
#pragma unroll
    for (int j = 0; j < 4; ++j) {
        const int chunk = tid + j * 256;          // 1024 chunks of 4 floats
        const int r  = chunk >> 3;
        const int c4 = (chunk & 7) * 4;
        const float4 v = *(const float4*)(src + (size_t)r * ld + c4);
        v4h h;
        h[0] = (_Float16)v.x; h[1] = (_Float16)v.y;
        h[2] = (_Float16)v.z; h[3] = (_Float16)v.w;
        *(v4h*)(dst + r * LDS_STRIDE + c4) = h;
    }
}

// Transpose-stage a 32x128 f16 tile: dst[n][k] = src[k0+k][n0+n].
__device__ __forceinline__ void stage_vT(const _Float16* __restrict__ Vb, int k0, int n0,
                                         int C, _Float16* dst, int tid) {
#pragma unroll
    for (int j = 0; j < 2; ++j) {
        const int chunk = tid + j * 256;          // 512 chunks of 8 halves along n
        const int k  = chunk >> 4;
        const int n8 = (chunk & 15) * 8;
        const v8h v = *(const v8h*)(Vb + (size_t)(k0 + k) * C + n0 + n8);
#pragma unroll
        for (int e = 0; e < 8; ++e)
            dst[(n8 + e) * LDS_STRIDE + k] = v[e];
    }
}

// ---------------------------------------------------------------------------
// Fragment loaders (CDNA5 ISA 7.12.2 layouts, wave32).
// ---------------------------------------------------------------------------
__device__ __forceinline__ v16h load_a_frag(const _Float16* base, int row0, int lane) {
    const int m = row0 + (lane & 15);
    const int g = (lane >> 4) & 1;
    const _Float16* p = base + m * LDS_STRIDE;
    v8h lo = *(const v8h*)(p + g * 8);
    v8h hi = *(const v8h*)(p + 16 + g * 8);
    v16h a;
#pragma unroll
    for (int i = 0; i < 8; ++i) { a[i] = lo[i]; a[8 + i] = hi[i]; }
    return a;
}

__device__ __forceinline__ v16h load_b_frag(const _Float16* base, int col0, int lane) {
    const int n = col0 + (lane & 15);
    const int g = (lane >> 4) & 1;
    const _Float16* p = base + n * LDS_STRIDE;
    v8h lo = *(const v8h*)(p + g * 8);
    v8h hi = *(const v8h*)(p + 16 + g * 8);
    v16h b;
#pragma unroll
    for (int i = 0; i < 8; ++i) { b[i] = lo[i]; b[8 + i] = hi[i]; }
    return b;
}

// One 16x128 wave-tile k-step: A frag + 8 B frags, dense 8-WMMA burst.
__device__ __forceinline__ void mma_tile(const _Float16* As, const _Float16* Bs,
                                         int wave, int lane, v8f* acc) {
    const v16h a = load_a_frag(As, wave * 16, lane);
    v16h b[8];
#pragma unroll
    for (int t = 0; t < 8; ++t) b[t] = load_b_frag(Bs, t * 16, lane);
#pragma unroll
    for (int t = 0; t < 8; ++t)
        acc[t] = __builtin_amdgcn_wmma_f32_16x16x32_f16(
            false, a, false, b[t], (short)0, acc[t], false, false);
}

// ---------------------------------------------------------------------------
// Kernel 1: Y[m,n] = sum_k X[m,k]*W[n,k]  (f32 in, f16 out)  "NT" GEMM
// ---------------------------------------------------------------------------
__global__ __launch_bounds__(256) void proj_f32_to_f16(
    const float* __restrict__ X, const float* __restrict__ W,
    _Float16* __restrict__ Y, int M, int N, int K)
{
    __shared__ __align__(16) _Float16 As[TILE_HALFS];
    __shared__ __align__(16) _Float16 Bs[TILE_HALFS];
    const int tid  = threadIdx.x;
    const int lane = tid & 31;
    const int wave = tid >> 5;
    const int m0 = blockIdx.x * 128;
    const int n0 = blockIdx.y * 128;

    v8f acc[8] = {};

    for (int k0 = 0; k0 < K; k0 += 32) {
        stage_f32_cvt(X + (size_t)m0 * K + k0, K, As, tid);
        stage_f32_cvt(W + (size_t)n0 * K + k0, K, Bs, tid);
        __syncthreads();
        mma_tile(As, Bs, wave, lane, acc);
        __syncthreads();
    }

    const int g  = (lane >> 4) & 1;
    const int nl = lane & 15;
#pragma unroll
    for (int t = 0; t < 8; ++t)
#pragma unroll
        for (int i = 0; i < 8; ++i)
            Y[(size_t)(m0 + wave * 16 + i + 8 * g) * N + n0 + t * 16 + nl] =
                (_Float16)acc[t][i];
}

// ---------------------------------------------------------------------------
// Kernel 2: S[q,k] = scale * sum_c Q[q,c]*K[k,c]   (f16, "NT")
// Tiles staged by the Tensor Data Mover (wave 0 issues, TENSORcnt drains),
// double-buffered so the DMA for step kt+1 overlaps the WMMAs of step kt.
// ---------------------------------------------------------------------------
__global__ __launch_bounds__(256) void scores_qk(
    const _Float16* __restrict__ Q, const _Float16* __restrict__ Km,
    _Float16* __restrict__ S, int L, int C, float scale)
{
    __shared__ __align__(16) _Float16 As[2][TILE_HALFS];
    __shared__ __align__(16) _Float16 Bs[2][TILE_HALFS];
    const int tid  = threadIdx.x;
    const int lane = tid & 31;
    const int wave = tid >> 5;
    const int m0 = blockIdx.x * 128;
    const int n0 = blockIdx.y * 128;

    const size_t zoff = (size_t)blockIdx.z * L * C;
    const _Float16* Qb = Q  + zoff + (size_t)m0 * C;
    const _Float16* Kb = Km + zoff + (size_t)n0 * C;
    _Float16*       Sb = S  + (size_t)blockIdx.z * L * L;

    const unsigned ldsA[2] = { lds_offset_of(As[0]), lds_offset_of(As[1]) };
    const unsigned ldsB[2] = { lds_offset_of(Bs[0]), lds_offset_of(Bs[1]) };
    const bool producer = (wave == 0);

    v8f acc[8] = {};
    const int KT = C / 32;

    if (producer) {
        tdm_load_tile_f16(Qb, ldsA[0], (unsigned)C);
        tdm_load_tile_f16(Kb, ldsB[0], (unsigned)C);
        wait_tensor_all();
    }
    __syncthreads();

    for (int kt = 0; kt < KT; ++kt) {
        const int cur = kt & 1;
        if (producer && (kt + 1 < KT)) {
            tdm_load_tile_f16(Qb + (kt + 1) * 32, ldsA[cur ^ 1], (unsigned)C);
            tdm_load_tile_f16(Kb + (kt + 1) * 32, ldsB[cur ^ 1], (unsigned)C);
        }
        mma_tile(As[cur], Bs[cur], wave, lane, acc);
        if (producer) wait_tensor_all();
        __syncthreads();
    }

    const int g  = (lane >> 4) & 1;
    const int nl = lane & 15;
#pragma unroll
    for (int t = 0; t < 8; ++t)
#pragma unroll
        for (int i = 0; i < 8; ++i)
            Sb[(size_t)(m0 + wave * 16 + i + 8 * g) * L + n0 + t * 16 + nl] =
                (_Float16)(acc[t][i] * scale);
}

// ---------------------------------------------------------------------------
// Kernel 3: in-place row softmax over 4096 f16 values. One block per row.
// ---------------------------------------------------------------------------
__global__ __launch_bounds__(256) void softmax_rows(_Float16* __restrict__ S)
{
    __shared__ float red[256];
    const int tid = threadIdx.x;
    _Float16* p = S + (size_t)blockIdx.x * 4096;

    float vals[16];
    float m = -1e30f;
#pragma unroll
    for (int i = 0; i < 16; ++i) {
        float v = (float)p[tid + i * 256];
        vals[i] = v;
        m = fmaxf(m, v);
    }
    red[tid] = m; __syncthreads();
#pragma unroll
    for (int s = 128; s > 0; s >>= 1) {
        if (tid < s) red[tid] = fmaxf(red[tid], red[tid + s]);
        __syncthreads();
    }
    m = red[0]; __syncthreads();

    float sum = 0.f;
#pragma unroll
    for (int i = 0; i < 16; ++i) {
        vals[i] = __expf(vals[i] - m);
        sum += vals[i];
    }
    red[tid] = sum; __syncthreads();
#pragma unroll
    for (int s = 128; s > 0; s >>= 1) {
        if (tid < s) red[tid] += red[tid + s];
        __syncthreads();
    }
    const float inv = 1.f / red[0];
#pragma unroll
    for (int i = 0; i < 16; ++i)
        p[tid + i * 256] = (_Float16)(vals[i] * inv);
}

// ---------------------------------------------------------------------------
// Kernel 4: O[q,c] = sum_k P[q,k]*V[k,c]   (f16 in, f32 out, "NN",
// double-buffered; A staged async-to-LDS, V transposed through LDS)
// ---------------------------------------------------------------------------
__global__ __launch_bounds__(256) void attn_pv(
    const _Float16* __restrict__ P, const _Float16* __restrict__ V,
    float* __restrict__ O, int L, int C)
{
    __shared__ __align__(16) _Float16 As[2][TILE_HALFS];
    __shared__ __align__(16) _Float16 Vs[2][TILE_HALFS];
    const int tid  = threadIdx.x;
    const int lane = tid & 31;
    const int wave = tid >> 5;
    const int m0 = blockIdx.x * 128;
    const int n0 = blockIdx.y * 128;

    const _Float16* Pb = P + (size_t)blockIdx.z * L * L + (size_t)m0 * L;
    const _Float16* Vb = V + (size_t)blockIdx.z * L * C;
    float*          Ob = O + (size_t)blockIdx.z * L * C;

    v8f acc[8] = {};
    const int KT = L / 32;

    stage_f16(Pb, L, As[0], tid);
    stage_vT(Vb, 0, n0, C, Vs[0], tid);
    wait_async_all();
    __syncthreads();

    for (int kt = 0; kt < KT; ++kt) {
        const int cur = kt & 1;
        if (kt + 1 < KT) {
            stage_f16(Pb + (kt + 1) * 32, L, As[cur ^ 1], tid);
            stage_vT(Vb, (kt + 1) * 32, n0, C, Vs[cur ^ 1], tid);
        }
        mma_tile(As[cur], Vs[cur], wave, lane, acc);
        wait_async_all();
        __syncthreads();
    }

    const int g  = (lane >> 4) & 1;
    const int nl = lane & 15;
#pragma unroll
    for (int t = 0; t < 8; ++t)
#pragma unroll
        for (int i = 0; i < 8; ++i)
            Ob[(size_t)(m0 + wave * 16 + i + 8 * g) * C + n0 + t * 16 + nl] =
                acc[t][i];
}

// ---------------------------------------------------------------------------
// Launch: x[4,64,64,512] f32, wq/wk[512,512] f32 -> out f32.
// Workspace: Q16 (16MB) | K16 (16MB) | S/P f16 (128MB)
// ---------------------------------------------------------------------------
extern "C" void kernel_launch(void* const* d_in, const int* in_sizes, int n_in,
                              void* d_out, int out_size, void* d_ws, size_t ws_size,
                              hipStream_t stream) {
    (void)in_sizes; (void)n_in; (void)out_size; (void)ws_size;

    const float* x  = (const float*)d_in[0];
    const float* wq = (const float*)d_in[1];
    const float* wk = (const float*)d_in[2];
    float* out = (float*)d_out;

    const int NB = 4, L = 4096, C = 512;
    const int M = NB * L;  // 16384 rows across all batches

    char* ws = (char*)d_ws;
    const size_t qBytes = (size_t)M * C * sizeof(_Float16);  // 16 MB
    _Float16* Q16 = (_Float16*)(ws);
    _Float16* K16 = (_Float16*)(ws + qBytes);
    _Float16* S   = (_Float16*)(ws + 2 * qBytes);            // 128 MB f16 scores

    dim3 blk(256, 1, 1);

    proj_f32_to_f16<<<dim3(M / 128, C / 128, 1), blk, 0, stream>>>(x, wq, Q16, M, C, C);
    proj_f32_to_f16<<<dim3(M / 128, C / 128, 1), blk, 0, stream>>>(x, wk, K16, M, C, C);

    const float scale = 0.044194173824159216f;  // 1/sqrt(512)
    scores_qk<<<dim3(L / 128, L / 128, NB), blk, 0, stream>>>(Q16, K16, S, L, C, scale);

    softmax_rows<<<dim3(M, 1, 1), blk, 0, stream>>>(S);

    attn_pv<<<dim3(L / 128, C / 128, NB), blk, 0, stream>>>(S, K16, out, L, C);
}
